// uQDNN_ATTENTION_61100204753089
// MI455X (gfx1250) — matrix operand order repaired and checked by reading, not compiled, not run on hardware
//
#include <hip/hip_runtime.h>
#include <hip/hip_bf16.h>
#include <math.h>

typedef __attribute__((ext_vector_type(16))) _Float16 v16h;
typedef __attribute__((ext_vector_type(8)))  float    v8f;
typedef _Float16 h16;

#define B_   64
#define S_   128
#define V_   5000
#define C_   16
#define M_   4096
#define U_   256
#define H_   128
#define DT_  300
#define DA_  74
#define DV_  35
#define CELL_ 64

// ---------------------------------------------------------------------------
// WMMA fragment helpers (wave32, V_WMMA_F32_16X16X32_F16 operands; this file's own contiguous k order, not the ISA table order)
// A: 16x32 (MxK). lane l holds row M=l&15; lanes 0-15 hold K=0..15,
//    lanes 16-31 hold K=16..31 (same k order as B).
// B: 32x16 (KxN). lane l holds col N=l&15; lanes 0-15 K=0..15, lanes 16-31 K=16..31.
// Both loaders read only CONTIGUOUS runs of halves (merged into global_load_b128).
// ---------------------------------------------------------------------------
__device__ __forceinline__ v16h ld_a16(const h16* base, int ld, int row0, int k0) {
  int l = threadIdx.x & 31;
  const h16* p = base + (size_t)(row0 + (l & 15)) * ld + k0 + ((l >> 4) << 4);
  v16h a;
#pragma unroll
  for (int i = 0; i < 8; ++i) { a[i] = p[i]; a[i + 8] = p[8 + i]; }
  return a;
}

// B[k][n] = base[(n0+n)*ld + k0+k]  (K is the contiguous dimension of base)
__device__ __forceinline__ v16h ld_b16(const h16* base, int ld, int n0, int k0) {
  int l = threadIdx.x & 31;
  const h16* p = base + (size_t)(n0 + (l & 15)) * ld + k0 + ((l >> 4) << 4);
  v16h b;
#pragma unroll
  for (int i = 0; i < 16; ++i) b[i] = p[i];
  return b;
}

__device__ __forceinline__ v8f wmma_f16(v16h a, v16h b, v8f c) {
  return __builtin_amdgcn_wmma_f32_16x16x32_f16(false, a, false, b, (short)0, c,
                                                false, false);
}

// C/D tile store helpers (C layout: lane l covers col l&15; VGPR j covers row
// j + 8*(l>>4) per ISA table)
__device__ __forceinline__ void store_diff_sum(float* ur, float* ui, int ld,
                                               v8f P, v8f Q, v8f R, v8f S) {
  int l = threadIdx.x & 31;
  int col = l & 15, rb = (l >> 4) * 8;
#pragma unroll
  for (int j = 0; j < 8; ++j) {
    ur[(size_t)(j + rb) * ld + col] = P[j] - Q[j];
    ui[(size_t)(j + rb) * ld + col] = R[j] + S[j];
  }
}

// ---------------------------------------------------------------------------
// 1) Embedding lookup + LSTM over S steps + 128->16 projection (amp_text)
// ---------------------------------------------------------------------------
__global__ __launch_bounds__(512) void lstm_kernel(
    const int* __restrict__ text_idx, const float* __restrict__ embed,
    const float* __restrict__ Wih, const float* __restrict__ Whh,
    const float* __restrict__ bih, const float* __restrict__ bhh,
    const float* __restrict__ projW, const float* __restrict__ projB,
    float* __restrict__ amp_text) {
  __shared__ float xs[DT_];
  __shared__ float hs[H_];
  __shared__ float cs[H_];
  __shared__ float gs[4 * H_];
  int b = blockIdx.x, t = threadIdx.x;
  if (t < H_) { hs[t] = 0.f; cs[t] = 0.f; }
  __syncthreads();
  for (int s = 0; s < S_; ++s) {
    int idx = text_idx[b * S_ + s];
    for (int i = t; i < DT_; i += 512) xs[i] = embed[(size_t)idx * DT_ + i];
    __syncthreads();
    {
      const float* wi = Wih + (size_t)t * DT_;
      const float* wh = Whh + (size_t)t * H_;
      float acc = bih[t] + bhh[t];
      for (int d = 0; d < DT_; ++d) acc += wi[d] * xs[d];
      for (int k = 0; k < H_; ++k) acc += wh[k] * hs[k];
      gs[t] = acc;
    }
    __syncthreads();
    if (t < H_) {
      float ig = 1.f / (1.f + __expf(-gs[t]));
      float fg = 1.f / (1.f + __expf(-gs[H_ + t]));
      float gg = tanhf(gs[2 * H_ + t]);
      float og = 1.f / (1.f + __expf(-gs[3 * H_ + t]));
      float cn = fg * cs[t] + ig * gg;
      cs[t] = cn;
      hs[t] = og * tanhf(cn);
    }
    __syncthreads();
    if (t < C_) {
      float acc = projB[t];
      const float* pw = projW + t * H_;
      for (int k = 0; k < H_; ++k) acc += pw[k] * hs[k];
      amp_text[((size_t)b * S_ + s) * C_ + t] = acc;
    }
    __syncthreads();
  }
}

// ---------------------------------------------------------------------------
// 2) 3-layer ReLU MLP per token (audio & video), one thread per token
// ---------------------------------------------------------------------------
__global__ void mlp_kernel(const float* __restrict__ in, int din,
                           const float* __restrict__ W1, const float* __restrict__ b1,
                           const float* __restrict__ W2, const float* __restrict__ b2,
                           const float* __restrict__ W3, const float* __restrict__ b3,
                           float* __restrict__ out) {
  int tok = blockIdx.x * blockDim.x + threadIdx.x;
  if (tok >= B_ * S_) return;
  const float* xp = in + (size_t)tok * din;
  float y1[C_], y2[C_];
  for (int c = 0; c < C_; ++c) {
    float a = b1[c];
    const float* w = W1 + c * din;
    for (int d = 0; d < din; ++d) a += w[d] * xp[d];
    y1[c] = fmaxf(a, 0.f);
  }
  for (int c = 0; c < C_; ++c) {
    float a = b2[c];
    for (int d = 0; d < C_; ++d) a += W2[c * C_ + d] * y1[d];
    y2[c] = fmaxf(a, 0.f);
  }
  for (int c = 0; c < C_; ++c) {
    float a = b3[c];
    for (int d = 0; d < C_; ++d) a += W3[c * C_ + d] * y2[d];
    out[(size_t)tok * C_ + c] = fmaxf(a, 0.f);
  }
}

// ---------------------------------------------------------------------------
// 3) Sequence-softmax of per-modality norms -> combined weight (B,S)
// ---------------------------------------------------------------------------
__global__ __launch_bounds__(128) void weight_kernel(
    const float* amp_t, const float* amp_a, const float* amp_v,
    const float* mweights, float* __restrict__ weight) {
  __shared__ float red[S_];
  __shared__ float sm[3][S_];
  int b = blockIdx.x, s = threadIdx.x;
  const float* amps[3] = {amp_t, amp_a, amp_v};
  float norms[3];
  for (int i = 0; i < 3; ++i) {
    const float* a = amps[i] + ((size_t)b * S_ + s) * C_;
    float ss = 0.f;
    for (int c = 0; c < C_; ++c) ss += a[c] * a[c];
    norms[i] = sqrtf(ss);
  }
  for (int i = 0; i < 3; ++i) {
    red[s] = norms[i]; __syncthreads();
    for (int o = 64; o > 0; o >>= 1) { if (s < o) red[s] = fmaxf(red[s], red[s + o]); __syncthreads(); }
    float mx = red[0]; __syncthreads();
    float e = __expf(norms[i] - mx);
    red[s] = e; __syncthreads();
    for (int o = 64; o > 0; o >>= 1) { if (s < o) red[s] += red[s + o]; __syncthreads(); }
    float sum = red[0]; __syncthreads();
    sm[i][s] = e / sum;
  }
  float m0 = mweights[0], m1 = mweights[1], m2 = mweights[2];
  float mm = fmaxf(m0, fmaxf(m1, m2));
  float e0 = __expf(m0 - mm), e1 = __expf(m1 - mm), e2 = __expf(m2 - mm);
  float es = e0 + e1 + e2;
  weight[b * S_ + s] = (e0 * sm[0][s] + e1 * sm[1][s] + e2 * sm[2][s]) / es;
}

// ---------------------------------------------------------------------------
// 4) Normalize amps, apply phases, 3-way complex Kronecker -> tr/ti (f16)
//    plus transposed copies trT/tiT (B,M,S) so WMMA B-loads stay contiguous.
// ---------------------------------------------------------------------------
__global__ __launch_bounds__(256) void pack_kernel(
    const int* __restrict__ text_idx, const float* __restrict__ phase_tables,
    const float* amp_t, const float* amp_a, const float* amp_v,
    h16* __restrict__ trh, h16* __restrict__ tih,
    h16* __restrict__ trT, h16* __restrict__ tiT) {
  __shared__ float hr[3][C_], hi[3][C_], rn[3], amp[3][C_];
  int tok = blockIdx.x;
  int b = tok >> 7, s = tok & 127;
  int t = threadIdx.x;
  int idx = text_idx[tok];
  if (t < 48) {
    int i = t >> 4, c = t & 15;
    const float* src = (i == 0) ? amp_t : (i == 1) ? amp_a : amp_v;
    amp[i][c] = src[(size_t)tok * C_ + c];
  }
  __syncthreads();
  if (t < 3) {
    float ss = 0.f;
    for (int c = 0; c < C_; ++c) ss += amp[t][c] * amp[t][c];
    rn[t] = rsqrtf(ss);
  }
  __syncthreads();
  if (t < 48) {
    int i = t >> 4, c = t & 15;
    float ph = phase_tables[((size_t)i * V_ + idx) * C_ + c];
    float an = amp[i][c] * rn[i];
    hr[i][c] = cosf(ph) * an;
    hi[i][c] = sinf(ph) * an;
  }
  __syncthreads();
  int a = t >> 4, d = t & 15;
  float tr1 = hr[0][a] - hi[0][a];               // (1+i)*(hr0 + i*hi0)
  float ti1 = hr[0][a] + hi[0][a];
  float tr2 = tr1 * hr[1][d] - ti1 * hi[1][d];
  float ti2 = ti1 * hr[1][d] + tr1 * hi[1][d];
  size_t base  = (size_t)tok * M_ + (size_t)t * 16;
  size_t tbase = ((size_t)b * M_ + (size_t)t * 16) * S_ + s;
#pragma unroll
  for (int v = 0; v < C_; ++v) {
    float r3 = tr2 * hr[2][v] - ti2 * hi[2][v];
    float i3 = ti2 * hr[2][v] + tr2 * hi[2][v];
    trh[base + v] = (h16)r3;
    tih[base + v] = (h16)i3;
    trT[tbase + (size_t)v * S_] = (h16)r3;
    tiT[tbase + (size_t)v * S_] = (h16)i3;
  }
}

// ---------------------------------------------------------------------------
// 5) Complex attention scores: Ur = tr*tr^T - ti*ti^T, Ui = ti*tr^T + tr*ti^T
//    One wave per 32x32 output region (2x2 register blocking): 8 fragment
//    loads feed 16 WMMAs per K-step -> 2x L2 arithmetic intensity.
// ---------------------------------------------------------------------------
__global__ __launch_bounds__(32) void scores_kernel(
    const h16* __restrict__ trh, const h16* __restrict__ tih,
    float* __restrict__ Ur, float* __restrict__ Ui) {
  int b = blockIdx.x, s0 = blockIdx.y * 32, t0 = blockIdx.z * 32;
  const h16* trb = trh + (size_t)b * S_ * M_;
  const h16* tib = tih + (size_t)b * S_ * M_;
  v8f P00 = {}, Q00 = {}, R00 = {}, S00 = {};
  v8f P01 = {}, Q01 = {}, R01 = {}, S01 = {};
  v8f P10 = {}, Q10 = {}, R10 = {}, S10 = {};
  v8f P11 = {}, Q11 = {}, R11 = {}, S11 = {};
  for (int k0 = 0; k0 < M_; k0 += 32) {
    v16h fr0 = ld_a16(trb, M_, s0,      k0);
    v16h fi0 = ld_a16(tib, M_, s0,      k0);
    v16h fr1 = ld_a16(trb, M_, s0 + 16, k0);
    v16h fi1 = ld_a16(tib, M_, s0 + 16, k0);
    v16h br0 = ld_b16(trb, M_, t0,      k0);  // B[k][t] = tr[t][k]
    v16h bi0 = ld_b16(tib, M_, t0,      k0);
    v16h br1 = ld_b16(trb, M_, t0 + 16, k0);
    v16h bi1 = ld_b16(tib, M_, t0 + 16, k0);
    P00 = wmma_f16(fr0, br0, P00); Q00 = wmma_f16(fi0, bi0, Q00);
    R00 = wmma_f16(fi0, br0, R00); S00 = wmma_f16(fr0, bi0, S00);
    P01 = wmma_f16(fr0, br1, P01); Q01 = wmma_f16(fi0, bi1, Q01);
    R01 = wmma_f16(fi0, br1, R01); S01 = wmma_f16(fr0, bi1, S01);
    P10 = wmma_f16(fr1, br0, P10); Q10 = wmma_f16(fi1, bi0, Q10);
    R10 = wmma_f16(fi1, br0, R10); S10 = wmma_f16(fr1, bi0, S10);
    P11 = wmma_f16(fr1, br1, P11); Q11 = wmma_f16(fi1, bi1, Q11);
    R11 = wmma_f16(fi1, br1, R11); S11 = wmma_f16(fr1, bi1, S11);
  }
  float* urb = Ur + ((size_t)b * S_ + s0) * S_ + t0;
  float* uib = Ui + ((size_t)b * S_ + s0) * S_ + t0;
  store_diff_sum(urb,                    uib,                    S_, P00, Q00, R00, S00);
  store_diff_sum(urb + 16,               uib + 16,               S_, P01, Q01, R01, S01);
  store_diff_sum(urb + (size_t)16 * S_,  uib + (size_t)16 * S_,  S_, P10, Q10, R10, S10);
  store_diff_sum(urb + (size_t)16 * S_ + 16, uib + (size_t)16 * S_ + 16, S_, P11, Q11, R11, S11);
}

// ---------------------------------------------------------------------------
// 6) Row softmax over t for Ur and Ui -> f16 attention probs
// ---------------------------------------------------------------------------
__global__ __launch_bounds__(128) void softmax_kernel(
    const float* __restrict__ Ur, const float* __restrict__ Ui,
    h16* __restrict__ arh, h16* __restrict__ aih) {
  __shared__ float red[S_];
  int row = blockIdx.x;                  // 0 .. 2*B*S-1
  int z = row >= B_ * S_;
  int r = z ? row - B_ * S_ : row;
  const float* in = (z ? Ui : Ur) + (size_t)r * S_;
  h16* out = (z ? aih : arh) + (size_t)r * S_;
  int t = threadIdx.x;
  float v = in[t];
  red[t] = v; __syncthreads();
  for (int o = 64; o > 0; o >>= 1) { if (t < o) red[t] = fmaxf(red[t], red[t + o]); __syncthreads(); }
  float mx = red[0]; __syncthreads();
  float e = __expf(v - mx);
  red[t] = e; __syncthreads();
  for (int o = 64; o > 0; o >>= 1) { if (t < o) red[t] += red[t + o]; __syncthreads(); }
  out[t] = (h16)(e / red[0]);
}

// ---------------------------------------------------------------------------
// 7) Attention output + residual, IN PLACE over tr/ti (2x2 blocked):
//    out_r = ar@tr - ai@ti + tr ; out_i = ai@tr + ar@ti + ti.   K = S = 128.
// ---------------------------------------------------------------------------
__global__ __launch_bounds__(32) void attnout_kernel(
    const h16* __restrict__ arh, const h16* __restrict__ aih,
    const h16* __restrict__ trT, const h16* __restrict__ tiT,
    h16* trh, h16* tih) {
  int b = blockIdx.x, s0 = blockIdx.y * 32, d0 = blockIdx.z * 32;
  const h16* arb  = arh + (size_t)b * S_ * S_;
  const h16* aib  = aih + (size_t)b * S_ * S_;
  const h16* trTb = trT + (size_t)b * M_ * S_;
  const h16* tiTb = tiT + (size_t)b * M_ * S_;
  v8f A00 = {}, B00 = {}, C00 = {}, D00 = {};
  v8f A01 = {}, B01 = {}, C01 = {}, D01 = {};
  v8f A10 = {}, B10 = {}, C10 = {}, D10 = {};
  v8f A11 = {}, B11 = {}, C11 = {}, D11 = {};
#pragma unroll
  for (int k0 = 0; k0 < S_; k0 += 32) {
    v16h fr0 = ld_a16(arb, S_, s0,      k0);
    v16h fi0 = ld_a16(aib, S_, s0,      k0);
    v16h fr1 = ld_a16(arb, S_, s0 + 16, k0);
    v16h fi1 = ld_a16(aib, S_, s0 + 16, k0);
    v16h br0 = ld_b16(trTb, S_, d0,      k0);  // B[t][d] via transposed copy
    v16h bi0 = ld_b16(tiTb, S_, d0,      k0);
    v16h br1 = ld_b16(trTb, S_, d0 + 16, k0);
    v16h bi1 = ld_b16(tiTb, S_, d0 + 16, k0);
    A00 = wmma_f16(fr0, br0, A00); B00 = wmma_f16(fi0, bi0, B00);
    C00 = wmma_f16(fi0, br0, C00); D00 = wmma_f16(fr0, bi0, D00);
    A01 = wmma_f16(fr0, br1, A01); B01 = wmma_f16(fi0, bi1, B01);
    C01 = wmma_f16(fi0, br1, C01); D01 = wmma_f16(fr0, bi1, D01);
    A10 = wmma_f16(fr1, br0, A10); B10 = wmma_f16(fi1, bi0, B10);
    C10 = wmma_f16(fi1, br0, C10); D10 = wmma_f16(fr1, bi0, D10);
    A11 = wmma_f16(fr1, br1, A11); B11 = wmma_f16(fi1, bi1, B11);
    C11 = wmma_f16(fi1, br1, C11); D11 = wmma_f16(fr1, bi1, D11);
  }
  int l = threadIdx.x;
  int col = l & 15, rb = (l >> 4) * 8;
#pragma unroll
  for (int st = 0; st < 2; ++st) {
#pragma unroll
    for (int dt = 0; dt < 2; ++dt) {
      v8f a1 = st ? (dt ? A11 : A10) : (dt ? A01 : A00);
      v8f a2 = st ? (dt ? B11 : B10) : (dt ? B01 : B00);
      v8f a3 = st ? (dt ? C11 : C10) : (dt ? C01 : C00);
      v8f a4 = st ? (dt ? D11 : D10) : (dt ? D01 : D00);
#pragma unroll
      for (int j = 0; j < 8; ++j) {
        int srow = s0 + st * 16 + j + rb;
        size_t o = ((size_t)b * S_ + srow) * M_ + d0 + dt * 16 + col;
        float rres = (float)trh[o], ires = (float)tih[o];
        trh[o] = (h16)(a1[j] - a2[j] + rres);
        tih[o] = (h16)(a3[j] + a4[j] + ires);
      }
    }
  }
}

// ---------------------------------------------------------------------------
// 8) Normalize measurement projectors to f16
// ---------------------------------------------------------------------------
__global__ __launch_bounds__(128) void knorm_kernel(
    const float* __restrict__ kr, const float* __restrict__ ki,
    h16* __restrict__ krn, h16* __restrict__ kin) {
  __shared__ float red[128];
  int u = blockIdx.x, t = threadIdx.x;
  const float* r = kr + (size_t)u * M_;
  const float* im = ki + (size_t)u * M_;
  float ss = 0.f;
  for (int d = t; d < M_; d += 128) ss += r[d] * r[d] + im[d] * im[d];
  red[t] = ss; __syncthreads();
  for (int o = 64; o > 0; o >>= 1) { if (t < o) red[t] += red[t + o]; __syncthreads(); }
  float rn = rsqrtf(red[0]);
  for (int d = t; d < M_; d += 128) {
    krn[(size_t)u * M_ + d] = (h16)(r[d] * rn);
    kin[(size_t)u * M_ + d] = (h16)(im[d] * rn);
  }
}

// ---------------------------------------------------------------------------
// 9) Measurement GEMM (K=4096, 2x2 blocked) fused with |.|^2 and weighting.
//    inr = out_r@kr^T + out_i@ki^T ; ini = out_i@kr^T - out_r@ki^T.
// ---------------------------------------------------------------------------
__global__ __launch_bounds__(32) void measure_kernel(
    const h16* __restrict__ outr, const h16* __restrict__ outi,
    const h16* __restrict__ krn, const h16* __restrict__ kin,
    const float* __restrict__ weight, float* __restrict__ wprob) {
  int b = blockIdx.x, s0 = blockIdx.y * 32, u0 = blockIdx.z * 32;
  const h16* orb = outr + (size_t)b * S_ * M_;
  const h16* oib = outi + (size_t)b * S_ * M_;
  v8f RR00 = {}, II00 = {}, IR00 = {}, RI00 = {};
  v8f RR01 = {}, II01 = {}, IR01 = {}, RI01 = {};
  v8f RR10 = {}, II10 = {}, IR10 = {}, RI10 = {};
  v8f RR11 = {}, II11 = {}, IR11 = {}, RI11 = {};
  for (int k0 = 0; k0 < M_; k0 += 32) {
    v16h fr0 = ld_a16(orb, M_, s0,      k0);
    v16h fi0 = ld_a16(oib, M_, s0,      k0);
    v16h fr1 = ld_a16(orb, M_, s0 + 16, k0);
    v16h fi1 = ld_a16(oib, M_, s0 + 16, k0);
    v16h br0 = ld_b16(krn, M_, u0,      k0);   // B[k][u] = kr[u][k]
    v16h bi0 = ld_b16(kin, M_, u0,      k0);
    v16h br1 = ld_b16(krn, M_, u0 + 16, k0);
    v16h bi1 = ld_b16(kin, M_, u0 + 16, k0);
    RR00 = wmma_f16(fr0, br0, RR00); II00 = wmma_f16(fi0, bi0, II00);
    IR00 = wmma_f16(fi0, br0, IR00); RI00 = wmma_f16(fr0, bi0, RI00);
    RR01 = wmma_f16(fr0, br1, RR01); II01 = wmma_f16(fi0, bi1, II01);
    IR01 = wmma_f16(fi0, br1, IR01); RI01 = wmma_f16(fr0, bi1, RI01);
    RR10 = wmma_f16(fr1, br0, RR10); II10 = wmma_f16(fi1, bi0, II10);
    IR10 = wmma_f16(fi1, br0, IR10); RI10 = wmma_f16(fr1, bi0, RI10);
    RR11 = wmma_f16(fr1, br1, RR11); II11 = wmma_f16(fi1, bi1, II11);
    IR11 = wmma_f16(fi1, br1, IR11); RI11 = wmma_f16(fr1, bi1, RI11);
  }
  int l = threadIdx.x;
  int col = l & 15, rb = (l >> 4) * 8;
#pragma unroll
  for (int st = 0; st < 2; ++st) {
#pragma unroll
    for (int ut = 0; ut < 2; ++ut) {
      v8f rr = st ? (ut ? RR11 : RR10) : (ut ? RR01 : RR00);
      v8f ii = st ? (ut ? II11 : II10) : (ut ? II01 : II00);
      v8f ir = st ? (ut ? IR11 : IR10) : (ut ? IR01 : IR00);
      v8f ri = st ? (ut ? RI11 : RI10) : (ut ? RI01 : RI00);
#pragma unroll
      for (int j = 0; j < 8; ++j) {
        int s = s0 + st * 16 + j + rb;
        float inr = rr[j] + ii[j];
        float ini = ir[j] - ri[j];
        float prob = inr * inr + ini * ini;
        wprob[((size_t)b * S_ + s) * U_ + u0 + ut * 16 + col] =
            weight[b * S_ + s] * prob;
      }
    }
  }
}

// ---------------------------------------------------------------------------
// 10) Reduce over S + 3-layer head -> (B,1)
// ---------------------------------------------------------------------------
__global__ __launch_bounds__(256) void head_kernel(
    const float* __restrict__ wprob,
    const float* fc1W, const float* fc1b,
    const float* fc2W, const float* fc2b,
    const float* fc3W, const float* fc3b,
    float* __restrict__ out) {
  __shared__ float meas[U_];
  __shared__ float y1[CELL_];
  __shared__ float y2[CELL_];
  int b = blockIdx.x, t = threadIdx.x;
  float acc = 0.f;
  for (int s = 0; s < S_; ++s) acc += wprob[((size_t)b * S_ + s) * U_ + t];
  meas[t] = acc;
  __syncthreads();
  if (t < CELL_) {
    float a = fc1b[t];
    for (int u = 0; u < U_; ++u) a += fc1W[t * U_ + u] * meas[u];
    y1[t] = fmaxf(a, 0.f);
  }
  __syncthreads();
  if (t < CELL_) {
    float a = fc2b[t];
    for (int u = 0; u < CELL_; ++u) a += fc2W[t * CELL_ + u] * y1[u];
    y2[t] = fmaxf(a, 0.f);
  }
  __syncthreads();
  if (t == 0) {
    float a = fc3b[0];
    for (int u = 0; u < CELL_; ++u) a += fc3W[u] * y2[u];
    out[b] = a;
  }
}

// ---------------------------------------------------------------------------
extern "C" void kernel_launch(void* const* d_in, const int* in_sizes, int n_in,
                              void* d_out, int out_size, void* d_ws, size_t ws_size,
                              hipStream_t stream) {
  const int*   text_idx = (const int*)  d_in[0];
  const float* audio    = (const float*)d_in[1];
  const float* video    = (const float*)d_in[2];
  const float* embed    = (const float*)d_in[3];
  const float* phases   = (const float*)d_in[4];
  const float* lstm_Wih = (const float*)d_in[5];
  const float* lstm_Whh = (const float*)d_in[6];
  const float* lstm_bih = (const float*)d_in[7];
  const float* lstm_bhh = (const float*)d_in[8];
  const float* projW    = (const float*)d_in[9];
  const float* projB    = (const float*)d_in[10];
  const float* aW1 = (const float*)d_in[11]; const float* ab1 = (const float*)d_in[12];
  const float* aW2 = (const float*)d_in[13]; const float* ab2 = (const float*)d_in[14];
  const float* aW3 = (const float*)d_in[15]; const float* ab3 = (const float*)d_in[16];
  const float* vW1 = (const float*)d_in[17]; const float* vb1 = (const float*)d_in[18];
  const float* vW2 = (const float*)d_in[19]; const float* vb2 = (const float*)d_in[20];
  const float* vW3 = (const float*)d_in[21]; const float* vb3 = (const float*)d_in[22];
  const float* mweights = (const float*)d_in[23];
  const float* meas_kr  = (const float*)d_in[24];
  const float* meas_ki  = (const float*)d_in[25];
  const float* fc1W = (const float*)d_in[26]; const float* fc1b = (const float*)d_in[27];
  const float* fc2W = (const float*)d_in[28]; const float* fc2b = (const float*)d_in[29];
  const float* fc3W = (const float*)d_in[30]; const float* fc3b = (const float*)d_in[31];
  float* out = (float*)d_out;

  // workspace bump allocator (256B aligned)
  char* wp = (char*)d_ws;
  auto alloc = [&](size_t bytes) {
    char* p = wp;
    wp += (bytes + 255) & ~(size_t)255;
    return (void*)p;
  };
  float* amp_text = (float*)alloc((size_t)B_ * S_ * C_ * 4);
  float* amp_aud  = (float*)alloc((size_t)B_ * S_ * C_ * 4);
  float* amp_vid  = (float*)alloc((size_t)B_ * S_ * C_ * 4);
  float* weight   = (float*)alloc((size_t)B_ * S_ * 4);
  h16*   trh      = (h16*)  alloc((size_t)B_ * S_ * M_ * 2);
  h16*   tih      = (h16*)  alloc((size_t)B_ * S_ * M_ * 2);
  h16*   trT      = (h16*)  alloc((size_t)B_ * M_ * S_ * 2);
  h16*   tiT      = (h16*)  alloc((size_t)B_ * M_ * S_ * 2);
  float* Ur       = (float*)alloc((size_t)B_ * S_ * S_ * 4);
  float* Ui       = (float*)alloc((size_t)B_ * S_ * S_ * 4);
  h16*   arh      = (h16*)  alloc((size_t)B_ * S_ * S_ * 2);
  h16*   aih      = (h16*)  alloc((size_t)B_ * S_ * S_ * 2);
  h16*   krn      = (h16*)  alloc((size_t)U_ * M_ * 2);
  h16*   kin      = (h16*)  alloc((size_t)U_ * M_ * 2);
  float* wprob    = (float*)alloc((size_t)B_ * S_ * U_ * 4);

  // 1) LSTM + text projection
  lstm_kernel<<<B_, 512, 0, stream>>>(text_idx, embed, lstm_Wih, lstm_Whh,
                                      lstm_bih, lstm_bhh, projW, projB, amp_text);
  // 2) audio/video MLPs
  mlp_kernel<<<(B_ * S_ + 255) / 256, 256, 0, stream>>>(audio, DA_, aW1, ab1, aW2, ab2, aW3, ab3, amp_aud);
  mlp_kernel<<<(B_ * S_ + 255) / 256, 256, 0, stream>>>(video, DV_, vW1, vb1, vW2, vb2, vW3, vb3, amp_vid);
  // 3) sequence-softmax weights
  weight_kernel<<<B_, 128, 0, stream>>>(amp_text, amp_aud, amp_vid, mweights, weight);
  // 4) product states (f16 + transposed copies)
  pack_kernel<<<B_ * S_, 256, 0, stream>>>(text_idx, phases, amp_text, amp_aud, amp_vid,
                                           trh, tih, trT, tiT);
  // 5) attention scores (WMMA, K=4096, 32x32 per wave)
  scores_kernel<<<dim3(B_, S_ / 32, S_ / 32), 32, 0, stream>>>(trh, tih, Ur, Ui);
  // 6) softmax rows
  softmax_kernel<<<2 * B_ * S_, 128, 0, stream>>>(Ur, Ui, arh, aih);
  // 7) attention output + residual (WMMA, K=128), in place over trh/tih
  attnout_kernel<<<dim3(B_, S_ / 32, M_ / 32), 32, 0, stream>>>(arh, aih, trT, tiT, trh, tih);
  // 8) projector normalization
  knorm_kernel<<<U_, 128, 0, stream>>>(meas_kr, meas_ki, krn, kin);
  // 9) measurement (WMMA, K=4096, 32x32 per wave) + |.|^2 + weighting
  measure_kernel<<<dim3(B_, S_ / 32, U_ / 32), 32, 0, stream>>>(trh, tih, krn, kin, weight, wprob);
  // 10) reduce over S + head MLP
  head_kernel<<<B_, 256, 0, stream>>>(wprob, fc1W, fc1b, fc2W, fc2b, fc3W, fc3b, out);
}
